// DeepGraphConv_13056700580489
// MI455X (gfx1250) — matrix operand order repaired and checked by reading, not compiled, not run on hardware
//
#include <hip/hip_runtime.h>

#define NNODES 50000
#define NEDGES 400000
#define DIN    1024
#define DH     256

typedef __attribute__((ext_vector_type(16))) __bf16 v16bf;
typedef __attribute__((ext_vector_type(2)))  __bf16 bf16x2;
typedef __attribute__((ext_vector_type(8)))  float  v8f;

__device__ __forceinline__ unsigned short f2bf(float f) {
    union { float f; unsigned u; } x; x.f = f;
    unsigned r = x.u + 0x7FFFu + ((x.u >> 16) & 1u);   // round-to-nearest-even
    return (unsigned short)(r >> 16);
}
__device__ __forceinline__ float bf2f(unsigned short h) {
    union { float f; unsigned u; } x; x.u = ((unsigned)h) << 16;
    return x.f;
}
// pack two f32 -> two bf16 in one dword (native V_CVT_PK_BF16_F32 when available)
__device__ __forceinline__ unsigned pk2(float a, float b) {
#if __has_builtin(__builtin_amdgcn_cvt_pk_bf16_f32)
    union { bf16x2 v; unsigned u; } x;
    x.v = __builtin_amdgcn_cvt_pk_bf16_f32(a, b);
    return x.u;
#else
    return (unsigned)f2bf(a) | ((unsigned)f2bf(b) << 16);
#endif
}

// ---------------------------------------------------------------------------
// Pack fp32 weight W[Kin x Kout] (row major) into WMMA-B bf16 layout:
// tile (nt,kt): element e of lane L holds W[kt*32 + (e&7) + 8*(L>>4) + 16*(e>>3)][nt*16 + (L&15)]
// stored contiguously: Wp[((nt*nK + kt)*32 + lane)*16 + e]
// ---------------------------------------------------------------------------
__global__ void pack_w(const float* __restrict__ W, unsigned short* __restrict__ Wp,
                       int Kin, int Kout) {
    int idx = blockIdx.x * 256 + threadIdx.x;
    if (idx >= Kin * Kout) return;
    int nK   = Kin >> 5;
    int e    = idx & 15;
    int lane = (idx >> 4) & 31;
    int tile = idx >> 9;
    int kt   = tile % nK;
    int nt   = tile / nK;
    int K    = kt * 32 + (e & 7) + 8 * (lane >> 4) + 16 * (e >> 3);
    int N    = nt * 16 + (lane & 15);
    Wp[idx] = f2bf(W[(size_t)K * Kout + N]);
}

// ---------------------------------------------------------------------------
// C[N x 256] = A[N x KIN] @ Wp(bf16 packed B-layout), fp32 WMMA accumulation.
// 512 threads = 16 waves; wave w -> N-tile w; block -> one 16-row M-tile.
// ABF: A is bf16 row-major (b128 loads, no convert). else f32 (cvt_pk RNE).
// MODE: 0 none(no bias), 1 bias+relu, 2 bias+tanh, 3 bias+sigmoid.
// OBF: store bf16 (feeds next GEMM), else f32.
// ---------------------------------------------------------------------------
template<int KIN, int MODE, bool ABF, bool OBF>
__global__ __launch_bounds__(512)
void gemm_wmma(const void* __restrict__ actv, const unsigned short* __restrict__ Wp,
               void* __restrict__ outv, const float* __restrict__ bias) {
    constexpr int nK = KIN >> 5;
    const int lane = threadIdx.x & 31;
    const int wave = threadIdx.x >> 5;
    const int hh   = lane >> 4;
    const int l16  = lane & 15;
    const int mt   = blockIdx.x;
    const int nt   = wave;                         // 16 waves cover 256 cols

    const size_t rowbase = (size_t)(mt * 16 + l16) * KIN;
    const unsigned short* bp = Wp + ((size_t)(nt * nK) * 32 + lane) * 16;

    v8f acc = {0.f, 0.f, 0.f, 0.f, 0.f, 0.f, 0.f, 0.f};

#pragma unroll 4
    for (int kt = 0; kt < nK; ++kt) {
        union { v16bf v; unsigned w[8]; uint4 q[2]; } A;
        if (ABF) {
            const unsigned short* ap =
                (const unsigned short*)actv + rowbase + kt * 32 + hh * 8;
            __builtin_prefetch((const char*)ap + 512, 0, 0);   // global_prefetch_b8
            A.q[0] = *(const uint4*)(ap);          // e0..7  : K = k0+8h..+7
            A.q[1] = *(const uint4*)(ap + 16);     // e8..15 : K = k0+16+8h..+7
        } else {
            const float* ap = (const float*)actv + rowbase + kt * 32 + hh * 8;
            __builtin_prefetch((const char*)ap + 1024, 0, 0);  // global_prefetch_b8
            float4 a0 = *(const float4*)(ap);
            float4 a1 = *(const float4*)(ap + 4);
            float4 a2 = *(const float4*)(ap + 16);
            float4 a3 = *(const float4*)(ap + 20);
            A.w[0] = pk2(a0.x, a0.y);  A.w[1] = pk2(a0.z, a0.w);
            A.w[2] = pk2(a1.x, a1.y);  A.w[3] = pk2(a1.z, a1.w);
            A.w[4] = pk2(a2.x, a2.y);  A.w[5] = pk2(a2.z, a2.w);
            A.w[6] = pk2(a3.x, a3.y);  A.w[7] = pk2(a3.z, a3.w);
        }

        union { v16bf v; uint4 q[2]; } B;
        const uint4* bq = (const uint4*)(bp + (size_t)kt * 512);
        B.q[0] = bq[0];
        B.q[1] = bq[1];

        acc = __builtin_amdgcn_wmma_f32_16x16x32_bf16(
                  false, A.v, false, B.v, (short)0, acc, false, false);
    }

    const int col = nt * 16 + l16;
    const float bv = (MODE == 0) ? 0.f : bias[col];
    const size_t obase = (size_t)(mt * 16 + hh * 8) * DH + col;
#pragma unroll
    for (int r = 0; r < 8; ++r) {
        float v = (MODE == 0) ? acc[r] : acc[r] + bv;
        if (MODE == 1) v = fmaxf(v, 0.f);
        if (MODE == 2) {
#if __has_builtin(__builtin_amdgcn_tanhf)
            v = __builtin_amdgcn_tanhf(v);
#else
            v = tanhf(v);
#endif
        }
        if (MODE == 3) v = __builtin_amdgcn_rcpf(1.f + __expf(-v));
        if (OBF) ((unsigned short*)outv)[obase + (size_t)r * DH] = f2bf(v);
        else     ((float*)outv)[obase + (size_t)r * DH] = v;
    }
}

// ---------------------------------------------------------------------------
__global__ void zero_f(float* p, int n) {
    int i = blockIdx.x * 256 + threadIdx.x;
    if (i < n) p[i] = 0.f;
}

// agg[dst] += y[src], 256 floats per edge, 64 lanes/edge * float4
__global__ void scatter_add(const float* __restrict__ y, const long long* __restrict__ ei,
                            float* __restrict__ agg, int E) {
    int tid  = blockIdx.x * 256 + threadIdx.x;
    int edge = tid >> 6;
    if (edge >= E) return;
    int c = (tid & 63) * 4;
    long long s = ei[edge];
    long long d = ei[E + edge];
    float4 v = *(const float4*)(y + (size_t)s * DH + c);
    float* dp = agg + (size_t)d * DH + c;
    atomicAdd(dp + 0, v.x); atomicAdd(dp + 1, v.y);
    atomicAdd(dp + 2, v.z); atomicAdd(dp + 3, v.w);
}

// hb = bf16(relu(y + agg + b[col])), 2 elements/thread, packed b32 store
__global__ void add3_relu_bf16(const float* __restrict__ y, const float* __restrict__ agg,
                               const float* __restrict__ b, unsigned* __restrict__ hb2,
                               int total2) {
    int i = blockIdx.x * 256 + threadIdx.x;   // pair index
    if (i >= total2) return;
    int j = i * 2;
    float v0 = fmaxf(y[j] + agg[j] + b[j & (DH - 1)], 0.f);
    float v1 = fmaxf(y[j + 1] + agg[j + 1] + b[(j + 1) & (DH - 1)], 0.f);
    hb2[i] = pk2(v0, v1);
}

// A[n] = sum_k ya[n,k]*yb[n,k]*wc[k] + bc ; one wave32 per node, 8 nodes/block
__global__ void gated_score(const float* __restrict__ ya, const float* __restrict__ yb,
                            const float* __restrict__ wc, const float* __restrict__ bc,
                            float* __restrict__ Aout) {
    int wave = threadIdx.x >> 5, lane = threadIdx.x & 31;
    int n = blockIdx.x * 8 + wave;
    const float* pa = ya + (size_t)n * DH;
    const float* pb = yb + (size_t)n * DH;
    float s = 0.f;
    for (int k = lane; k < DH; k += 32) s += pa[k] * pb[k] * wc[k];
    for (int off = 16; off; off >>= 1) s += __shfl_xor(s, off, 32);
    if (lane == 0) Aout[n] = s + bc[0];
}

// stats[0]=max(A), stats[1]=sum exp(A-max); single 1024-thread block
__global__ void softmax_stats(const float* __restrict__ A, int n, float* __restrict__ stats) {
    __shared__ float red[1024];
    int t = threadIdx.x;
    float m = -3.402823466e38f;
    for (int i = t; i < n; i += 1024) m = fmaxf(m, A[i]);
    red[t] = m; __syncthreads();
    for (int s = 512; s; s >>= 1) { if (t < s) red[t] = fmaxf(red[t], red[t + s]); __syncthreads(); }
    float mx = red[0]; __syncthreads();
    float sum = 0.f;
    for (int i = t; i < n; i += 1024) sum += __expf(A[i] - mx);
    red[t] = sum; __syncthreads();
    for (int s = 512; s; s >>= 1) { if (t < s) red[t] += red[t + s]; __syncthreads(); }
    if (t == 0) { stats[0] = mx; stats[1] = red[0]; }
}

// hacc[k] += sum_n exp(A[n]-mx) * x3[n,k]  (x3 in bf16)
__global__ void weighted_sum(const float* __restrict__ A, const unsigned short* __restrict__ x3b,
                             const float* __restrict__ stats, float* __restrict__ hacc) {
    int t = threadIdx.x;
    int n0 = blockIdx.x * 196;
    int n1 = n0 + 196; if (n1 > NNODES) n1 = NNODES;
    float mx = stats[0];
    float acc = 0.f;
    for (int n = n0; n < n1; ++n)
        acc += __expf(A[n] - mx) * bf2f(x3b[(size_t)n * DH + t]);
    atomicAdd(&hacc[t], acc);
}

// h = hacc/sumexp; h2 = relu(h@Wr+br); logits = h2@Wcls+bcls
__global__ void final_head(const float* __restrict__ hacc, const float* __restrict__ stats,
                           const float* __restrict__ Wr, const float* __restrict__ br,
                           const float* __restrict__ Wcls, const float* __restrict__ bcls,
                           float* __restrict__ out) {
    __shared__ float sh[DH];
    __shared__ float sh2[DH];
    int t = threadIdx.x;
    float inv = __builtin_amdgcn_rcpf(stats[1]);
    sh[t] = hacc[t] * inv;
    __syncthreads();
    float v = 0.f;
    for (int k = 0; k < DH; ++k) v += sh[k] * Wr[(size_t)k * DH + t];
    sh2[t] = fmaxf(v + br[t], 0.f);
    __syncthreads();
    if (t < 2) {
        float lg = bcls[t];
        for (int j = 0; j < DH; ++j) lg += sh2[j] * Wcls[j * 2 + t];
        out[t]     = lg;
        out[2 + t] = __builtin_amdgcn_rcpf(1.f + __expf(-lg));
    }
}

// ---------------------------------------------------------------------------
extern "C" void kernel_launch(void* const* d_in, const int* in_sizes, int n_in,
                              void* d_out, int out_size, void* d_ws, size_t ws_size,
                              hipStream_t stream) {
    const float*     x  = (const float*)d_in[0];
    const long long* ei = (const long long*)d_in[1];
    const float *c1W1=(const float*)d_in[2],  *c1b1=(const float*)d_in[3];
    const float *c1W2=(const float*)d_in[4],  *c1b2=(const float*)d_in[5];
    const float *c2W1=(const float*)d_in[6],  *c2b1=(const float*)d_in[7];
    const float *c2W2=(const float*)d_in[8],  *c2b2=(const float*)d_in[9];
    const float *c3W1=(const float*)d_in[10], *c3b1=(const float*)d_in[11];
    const float *c3W2=(const float*)d_in[12], *c3b2=(const float*)d_in[13];
    const float *Wa=(const float*)d_in[14],   *ba=(const float*)d_in[15];
    const float *Wb=(const float*)d_in[16],   *bb=(const float*)d_in[17];
    const float *Wc=(const float*)d_in[18],   *bc=(const float*)d_in[19];
    const float *Wr=(const float*)d_in[20],   *br=(const float*)d_in[21];
    const float *Wcls=(const float*)d_in[22], *bcls=(const float*)d_in[23];
    (void)in_sizes; (void)n_in; (void)out_size; (void)ws_size;

    unsigned char* ws = (unsigned char*)d_ws;
    // packed bf16 weights (1.4 MB)
    unsigned short* P_c1W1 = (unsigned short*)(ws + 0);        // 1024x256 -> 512KB
    unsigned short* P_c1W2 = (unsigned short*)(ws + 524288);
    unsigned short* P_c2W1 = (unsigned short*)(ws + 655360);
    unsigned short* P_c2W2 = (unsigned short*)(ws + 786432);
    unsigned short* P_c3W1 = (unsigned short*)(ws + 917504);
    unsigned short* P_c3W2 = (unsigned short*)(ws + 1048576);
    unsigned short* P_Wa   = (unsigned short*)(ws + 1179648);
    unsigned short* P_Wb   = (unsigned short*)(ws + 1310720);
    // activation buffers
    const size_t NB  = (size_t)NNODES * DH * sizeof(float);    // 51.2 MB
    const size_t NBH = NB / 2;                                 // 25.6 MB (bf16)
    float*          bufY   = (float*)(ws + 2097152);
    float*          bufAgg = (float*)(ws + 2097152 + NB);
    unsigned short* bufHb  = (unsigned short*)(ws + 2097152 + 2 * NB);
    unsigned short* bufXb1 = (unsigned short*)(ws + 2097152 + 2 * NB + NBH);
    unsigned short* bufXb2 = (unsigned short*)(ws + 2097152 + 2 * NB + 2 * NBH);
    float*          stats  = (float*)(ws + 2097152 + 2 * NB + 3 * NBH);
    float*          hacc   = (float*)(ws + 2097152 + 2 * NB + 3 * NBH + 256);

    const int TOT = NNODES * DH;  // 12.8M

    // ---- pack weights (bf16, WMMA B-layout) ----
    pack_w<<<1024, 256, 0, stream>>>(c1W1, P_c1W1, DIN, DH);
    pack_w<<<256, 256, 0, stream>>>(c1W2, P_c1W2, DH, DH);
    pack_w<<<256, 256, 0, stream>>>(c2W1, P_c2W1, DH, DH);
    pack_w<<<256, 256, 0, stream>>>(c2W2, P_c2W2, DH, DH);
    pack_w<<<256, 256, 0, stream>>>(c3W1, P_c3W1, DH, DH);
    pack_w<<<256, 256, 0, stream>>>(c3W2, P_c3W2, DH, DH);
    pack_w<<<256, 256, 0, stream>>>(Wa, P_Wa, DH, DH);
    pack_w<<<256, 256, 0, stream>>>(Wb, P_Wb, DH, DH);

    const dim3 gG(NNODES / 16, 1);

    // ---- GIN layer 1: (x + A x)@W1 = y + A y,  y = x@W1 ----
    gemm_wmma<DIN, 0, false, false><<<gG, 512, 0, stream>>>(x, P_c1W1, bufY, nullptr);
    zero_f<<<TOT / 256, 256, 0, stream>>>(bufAgg, TOT);
    scatter_add<<<NEDGES / 4, 256, 0, stream>>>(bufY, ei, bufAgg, NEDGES);
    add3_relu_bf16<<<TOT / 512, 256, 0, stream>>>(bufY, bufAgg, c1b1, (unsigned*)bufHb, TOT / 2);
    gemm_wmma<DH, 1, true, true><<<gG, 512, 0, stream>>>(bufHb, P_c1W2, bufXb1, c1b2);

    // ---- GIN layer 2 ----
    gemm_wmma<DH, 0, true, false><<<gG, 512, 0, stream>>>(bufXb1, P_c2W1, bufY, nullptr);
    zero_f<<<TOT / 256, 256, 0, stream>>>(bufAgg, TOT);
    scatter_add<<<NEDGES / 4, 256, 0, stream>>>(bufY, ei, bufAgg, NEDGES);
    add3_relu_bf16<<<TOT / 512, 256, 0, stream>>>(bufY, bufAgg, c2b1, (unsigned*)bufHb, TOT / 2);
    gemm_wmma<DH, 1, true, true><<<gG, 512, 0, stream>>>(bufHb, P_c2W2, bufXb2, c2b2);

    // ---- GIN layer 3 ----
    gemm_wmma<DH, 0, true, false><<<gG, 512, 0, stream>>>(bufXb2, P_c3W1, bufY, nullptr);
    zero_f<<<TOT / 256, 256, 0, stream>>>(bufAgg, TOT);
    scatter_add<<<NEDGES / 4, 256, 0, stream>>>(bufY, ei, bufAgg, NEDGES);
    add3_relu_bf16<<<TOT / 512, 256, 0, stream>>>(bufY, bufAgg, c3b1, (unsigned*)bufHb, TOT / 2);
    gemm_wmma<DH, 1, true, true><<<gG, 512, 0, stream>>>(bufHb, P_c3W2, bufXb1, c3b2); // x3 bf16

    // ---- gated attention pooling ----
    gemm_wmma<DH, 2, true, false><<<gG, 512, 0, stream>>>(bufXb1, P_Wa, bufY, ba);    // tanh
    gemm_wmma<DH, 3, true, false><<<gG, 512, 0, stream>>>(bufXb1, P_Wb, bufAgg, bb);  // sigmoid

    float* Aout = (float*)d_out + 4;  // A_T written directly to output
    gated_score<<<NNODES / 8, 256, 0, stream>>>(bufY, bufAgg, Wc, bc, Aout);
    softmax_stats<<<1, 1024, 0, stream>>>(Aout, NNODES, stats);
    zero_f<<<1, 256, 0, stream>>>(hacc, DH);
    weighted_sum<<<256, 256, 0, stream>>>(Aout, bufXb1, stats, hacc);
    final_head<<<1, 256, 0, stream>>>(hacc, stats, Wr, br, Wcls, bcls, (float*)d_out);
}
